// GraphClassifier_42528766165147
// MI455X (gfx1250) — matrix operand order repaired and checked by reading, not compiled, not run on hardware
//
#include <hip/hip_runtime.h>
#include <hip/hip_bf16.h>

#define N_NODES  20000
#define N_EDGES  320000
#define N_GRAPHS 256
#define DCH      256
#define NLAYERS  4
#define KCAT     512
#define LDK      528   // padded LDS row stride (ushorts), 1056 B = multiple of 16

typedef __attribute__((ext_vector_type(16))) __bf16 v16bf;
typedef __attribute__((ext_vector_type(8)))  float  v8f;

union Frag { uint4 u[2]; v16bf v; };

__device__ __forceinline__ unsigned short f2bf(float f) {
  unsigned int u = __float_as_uint(f);
  u += 0x7FFFu + ((u >> 16) & 1u);   // round-to-nearest-even
  return (unsigned short)(u >> 16);
}

// ---------------- generic zero ----------------
__global__ void zero_kern(float* __restrict__ p, int n) {
  int i = blockIdx.x * blockDim.x + threadIdx.x;
  if (i < n) p[i] = 0.f;
}

// ---------------- weight fragment prep (once) ----------------
// wfrag layout: [l][s(16 ksteps)][tg(16 ntiles)][lane(32)][e(16)] bf16
// B-frag lane mapping: n = tg*16 + lane%16 ; k = s*32 + 16*(lane/16) + 2*(e/2) + (e&1)
__global__ void prep_wfrag(const float* __restrict__ Wl, const float* __restrict__ Wr,
                           unsigned short* __restrict__ wfrag) {
  const int idx  = blockIdx.x;           // 0..1023 = l*256 + s*16 + tg
  const int s    = (idx >> 4) & 15;
  const int lGrp = idx >> 8;
  const int tg   = idx & 15;
  const int lane = threadIdx.x;
  const int half = lane >> 4;
  const int n    = tg * 16 + (lane & 15);
  unsigned short outv[16];
#pragma unroll
  for (int e = 0; e < 16; ++e) {
    int v = e >> 1;
    int k = s * 32 + half * 16 + 2 * v + (e & 1);
    float w = (k < DCH) ? Wl[((size_t)lGrp * DCH + k) * DCH + n]
                        : Wr[((size_t)lGrp * DCH + (k - DCH)) * DCH + n];
    outv[e] = f2bf(w);
  }
  unsigned short* dst = wfrag + ((size_t)idx * 32 + lane) * 16;
#pragma unroll
  for (int e = 0; e < 16; ++e) dst[e] = outv[e];
}

// ---------------- activation init ----------------
__global__ void init_x(const float* __restrict__ x, float* __restrict__ xcur,
                       unsigned short* __restrict__ xcat) {
  int r = blockIdx.x, c = threadIdx.x;
  float v = x[(size_t)r * DCH + c];
  xcur[(size_t)r * DCH + c] = v;
  xcat[(size_t)r * KCAT + DCH + c] = f2bf(v);
}

// ---------------- degree ----------------
__global__ void deg_accum(const long long* __restrict__ ei, float* __restrict__ deg) {
  int e = blockIdx.x * blockDim.x + threadIdx.x;
  if (e < N_EDGES) atomicAdd(&deg[ei[N_EDGES + e]], 1.0f);
}
__global__ void finish_invdeg(float* __restrict__ deg, int n) {
  int i = blockIdx.x * blockDim.x + threadIdx.x;
  if (i < n) { float d = deg[i]; deg[i] = (d > 0.f) ? 1.f / d : 0.f; }
}

// ---------------- edge scatter (mean-agg numerator) ----------------
__global__ void scatter_add(const long long* __restrict__ ei, const float* __restrict__ x,
                            float* __restrict__ agg) {
  long long e = blockIdx.x;
  int c = threadIdx.x;
  long long s = ei[e];
  long long d = ei[N_EDGES + e];
  atomicAdd(&agg[d * DCH + c], x[s * DCH + c]);
}

// ---------------- agg*invdeg -> bf16 concat left half ----------------
__global__ void conv_agg(const float* __restrict__ agg, const float* __restrict__ invdeg,
                         unsigned short* __restrict__ xcat) {
  int r = blockIdx.x, c = threadIdx.x;
  xcat[(size_t)r * KCAT + c] = f2bf(agg[(size_t)r * DCH + c] * invdeg[r]);
}

// ---------------- fused WMMA GEMM + bias + BN partial sums ----------------
// h(20000x256) = xcat(20000x512,bf16) @ Wcat(512x256,bf16) + bias
// block: 256 thr = 8 waves = 2(M) x 4(Ngroups of 64 cols); 32x256 tile, K=512 in LDS
__global__ void __launch_bounds__(256)
gemm_wmma_bn(const unsigned short* __restrict__ xcat,
             const unsigned short* __restrict__ wfrag,   // this layer
             const float* __restrict__ bias,
             float* __restrict__ h,
             float* __restrict__ bnsum, float* __restrict__ bnsq) {
  __shared__ __align__(16) unsigned short As[32 * LDK];
  __shared__ float s_sum[DCH];
  __shared__ float s_sq[DCH];

  const int tid  = threadIdx.x;
  const int lane = tid & 31;
  const int wave = tid >> 5;
  const int mrow = wave & 1;        // which 16-row half
  const int ngrp = wave >> 1;       // which 64-col group
  const int half = lane >> 4;
  const int m    = lane & 15;
  const int row0 = blockIdx.x * 32;

  s_sum[tid] = 0.f;
  s_sq[tid]  = 0.f;

  // stage 32x512 bf16 A-tile: 2048 uint4, 8 per thread, fully coalesced
  const uint4* gsrc = reinterpret_cast<const uint4*>(xcat + (size_t)row0 * KCAT);
#pragma unroll
  for (int i = 0; i < 8; ++i) {
    int j = i * 256 + tid;          // uint4 index; covers ushort j*8
    int r = j >> 6;
    int k = (j & 63) * 8;
    *reinterpret_cast<uint4*>(&As[r * LDK + k]) = gsrc[j];
  }
  __syncthreads();

  v8f acc[4] = {};
  const unsigned short* arow = &As[(mrow * 16 + m) * LDK];

  for (int s = 0; s < 16; ++s) {
    Frag A;
    const uint4* ap = reinterpret_cast<const uint4*>(arow + s * 32 + half * 16);
    A.u[0] = ap[0];                 // k = s*32 + 16*half + [0..7]   (vgpr 0..3)
    A.u[1] = ap[1];                 // k = s*32 + 16*half + [8..15] (vgpr 4..7)
#pragma unroll
    for (int t = 0; t < 4; ++t) {
      const int tg = ngrp * 4 + t;
      Frag B;
      const uint4* bp = reinterpret_cast<const uint4*>(
          wfrag + ((size_t)(s * 16 + tg) * 32 + lane) * 16);
      B.u[0] = bp[0];
      B.u[1] = bp[1];
      acc[t] = __builtin_amdgcn_wmma_f32_16x16x32_bf16(
          false, A.v, false, B.v, (short)0, acc[t], false, false);
    }
  }

  // epilogue: bias, store h, per-channel sum/sumsq partials
#pragma unroll
  for (int t = 0; t < 4; ++t) {
    const int n = (ngrp * 4 + t) * 16 + m;
    const float b = bias[n];
    float lsum = 0.f, lsq = 0.f;
#pragma unroll
    for (int v = 0; v < 8; ++v) {
      float val = acc[t][v] + b;
      int gm = row0 + mrow * 16 + half * 8 + v;  // C/D layout: M = vgpr + 8*(lane/16)
      h[(size_t)gm * DCH + n] = val;
      lsum += val;
      lsq  += val * val;
    }
    atomicAdd(&s_sum[n], lsum);
    atomicAdd(&s_sq[n], lsq);
  }
  __syncthreads();
  atomicAdd(&bnsum[tid], s_sum[tid]);
  atomicAdd(&bnsq[tid],  s_sq[tid]);
}

// ---------------- BN finalize: per-channel scale/shift ----------------
__global__ void bn_final(const float* __restrict__ bnsum, const float* __restrict__ bnsq,
                         const float* __restrict__ gamma, const float* __restrict__ beta,
                         float* __restrict__ scale, float* __restrict__ shift) {
  int c = threadIdx.x;
  const float invN = 1.f / (float)N_NODES;
  float mu  = bnsum[c] * invN;
  float var = bnsq[c] * invN - mu * mu;
  float inv = rsqrtf(var + 1e-5f);
  float sc  = gamma[c] * inv;
  scale[c] = sc;
  shift[c] = beta[c] - mu * sc;
}

// ---------------- BN apply + ReLU, write f32 + bf16 copies ----------------
__global__ void bn_apply(const float* __restrict__ h, const float* __restrict__ scale,
                         const float* __restrict__ shift, float* __restrict__ xcur,
                         unsigned short* __restrict__ xcat) {
  int r = blockIdx.x, c = threadIdx.x;
  float v = fmaxf(h[(size_t)r * DCH + c] * scale[c] + shift[c], 0.f);
  xcur[(size_t)r * DCH + c] = v;
  xcat[(size_t)r * KCAT + DCH + c] = f2bf(v);
}

// ---------------- graph pooling ----------------
__global__ void pool_scatter(const long long* __restrict__ batch, const float* __restrict__ x,
                             float* __restrict__ pool, float* __restrict__ cnt) {
  int r = blockIdx.x, c = threadIdx.x;
  long long b = batch[r];
  atomicAdd(&pool[b * DCH + c], x[(size_t)r * DCH + c]);
  if (c == 0) atomicAdd(&cnt[b], 1.0f);
}

// ---------------- MLP head: 256 -> 128 -> 64 -> 10 ----------------
__global__ void __launch_bounds__(128)
mlp_head(const float* __restrict__ pool, const float* __restrict__ cnt,
         const float* __restrict__ w1, const float* __restrict__ b1,
         const float* __restrict__ w2, const float* __restrict__ b2,
         const float* __restrict__ w3, const float* __restrict__ b3,
         float* __restrict__ out) {
  __shared__ float g[DCH];
  __shared__ float h1[128];
  __shared__ float h2[64];
  int b = blockIdx.x, t = threadIdx.x;
  float ic = 1.f / fmaxf(cnt[b], 1.f);
  for (int i = t; i < DCH; i += 128) g[i] = pool[(size_t)b * DCH + i] * ic;
  __syncthreads();
  float a1 = b1[t];
  for (int k = 0; k < DCH; ++k) a1 += g[k] * w1[k * 128 + t];
  h1[t] = fmaxf(a1, 0.f);
  __syncthreads();
  if (t < 64) {
    float a2 = b2[t];
    for (int k = 0; k < 128; ++k) a2 += h1[k] * w2[k * 64 + t];
    h2[t] = fmaxf(a2, 0.f);
  }
  __syncthreads();
  if (t < 10) {
    float a3 = b3[t];
    for (int k = 0; k < 64; ++k) a3 += h2[k] * w3[k * 10 + t];
    out[b * 10 + t] = a3;
  }
}

// ---------------- launch ----------------
extern "C" void kernel_launch(void* const* d_in, const int* in_sizes, int n_in,
                              void* d_out, int out_size, void* d_ws, size_t ws_size,
                              hipStream_t stream) {
  const float*     x     = (const float*)d_in[0];
  const long long* ei    = (const long long*)d_in[1];   // int64 (2, N_EDGES)
  const long long* batch = (const long long*)d_in[2];   // int64 (N_NODES,)
  const float*     Wl    = (const float*)d_in[3];
  const float*     bl    = (const float*)d_in[4];
  const float*     Wr    = (const float*)d_in[5];
  const float*     gamma = (const float*)d_in[6];
  const float*     beta  = (const float*)d_in[7];
  const float*     fc1w  = (const float*)d_in[8];
  const float*     fc1b  = (const float*)d_in[9];
  const float*     fc2w  = (const float*)d_in[10];
  const float*     fc2b  = (const float*)d_in[11];
  const float*     fc3w  = (const float*)d_in[12];
  const float*     fc3b  = (const float*)d_in[13];
  float* out = (float*)d_out;

  char* ws = (char*)d_ws;
  float*          invdeg = (float*)(ws + 0);                 //  80 KB (padded)
  float*          agg    = (float*)(ws + 81920);             //  20.48 MB
  float*          hbuf   = (float*)(ws + 20561920);          //  20.48 MB
  float*          xcur   = (float*)(ws + 41041920);          //  20.48 MB
  unsigned short* xcat   = (unsigned short*)(ws + 61521920); //  20.48 MB  [node][512] bf16
  unsigned short* wfrag  = (unsigned short*)(ws + 82001920); //   1.05 MB
  float*          bnsum  = (float*)(ws + 83050496);          //  256
  float*          bnsq   = bnsum + 256;
  float*          scale  = bnsum + 512;
  float*          shift  = bnsum + 768;
  float*          pool   = (float*)(ws + 83054592);          //  256x256
  float*          cnt    = (float*)(ws + 83316736);          //  256 (contiguous after pool)

  prep_wfrag<<<NLAYERS * 16 * 16, 32, 0, stream>>>(Wl, Wr, wfrag);
  init_x<<<N_NODES, 256, 0, stream>>>(x, xcur, xcat);

  zero_kern<<<(N_NODES + 255) / 256, 256, 0, stream>>>(invdeg, N_NODES);
  deg_accum<<<(N_EDGES + 255) / 256, 256, 0, stream>>>(ei, invdeg);
  finish_invdeg<<<(N_NODES + 255) / 256, 256, 0, stream>>>(invdeg, N_NODES);

  for (int l = 0; l < NLAYERS; ++l) {
    zero_kern<<<(N_NODES * DCH + 255) / 256, 256, 0, stream>>>(agg, N_NODES * DCH);
    zero_kern<<<2, 256, 0, stream>>>(bnsum, 512);
    scatter_add<<<N_EDGES, 256, 0, stream>>>(ei, xcur, agg);
    conv_agg<<<N_NODES, 256, 0, stream>>>(agg, invdeg, xcat);
    gemm_wmma_bn<<<N_NODES / 32, 256, 0, stream>>>(
        xcat, wfrag + (size_t)l * 16 * 16 * 32 * 16, bl + l * DCH, hbuf, bnsum, bnsq);
    bn_final<<<1, 256, 0, stream>>>(bnsum, bnsq, gamma + l * DCH, beta + l * DCH, scale, shift);
    bn_apply<<<N_NODES, 256, 0, stream>>>(hbuf, scale, shift, xcur, xcat);
  }

  zero_kern<<<(65792 + 255) / 256, 256, 0, stream>>>(pool, 65792);  // pool + cnt
  pool_scatter<<<N_NODES, 256, 0, stream>>>(batch, xcur, pool, cnt);
  mlp_head<<<N_GRAPHS, 128, 0, stream>>>(pool, cnt, fc1w, fc1b, fc2w, fc2b, fc3w, fc3b, out);
}
